// QRNN_11364483465321
// MI455X (gfx1250) — matrix operand order repaired and checked by reading, not compiled, not run on hardware
//
#include <hip/hip_runtime.h>

// ---------------------------------------------------------------------------
// QRNN fused kernel for gfx1250 (MI455X):
//   gates via f16 WMMA (v_wmma_f32_16x16x32_f16), fp32 accumulate,
//   x staged to LDS via GLOBAL_LOAD_ASYNC_TO_LDS_B128 (ASYNCcnt),
//   sequential linear recurrence fused in the same kernel (h carried in regs).
// ---------------------------------------------------------------------------

typedef __attribute__((ext_vector_type(16))) _Float16 v16h;
typedef __attribute__((ext_vector_type(8)))  _Float16 v8h;
typedef __attribute__((ext_vector_type(8)))  float    v8f;

#define NB     16            // batch
#define NT     4096          // time
#define ND     512           // input dim
#define NU     512           // units
#define KTOT   1024          // W*D
#define NKT    32            // KTOT / 32
#define NUT    (NU / 16)     // 32 u-tiles of 16 units
#define CHUNK  64            // timesteps per chunk
#define MT     (CHUNK / 16)  // 4 M-tiles per chunk
#define NWAVES 2             // waves per block (each owns one 16-unit tile)
#define XROWS  (CHUNK + 1)   // rows t0-1 .. t0+CHUNK-1
#define XSTR   520           // padded f16 row stride (bank-conflict-free b128)
#define SSTR   17            // scan buffer row stride (floats)

#define WPACK_HALFS (NUT * NKT * 3 * 32 * 16)   // 1,572,864 halfs (3 MiB)

// ---------------------------------------------------------------------------
// Prologue 1: pack fp32 weights into the exact per-lane WMMA B-operand order.
//   idx = (((ut*NKT + kt)*3 + g)*32 + lane)*16 + e
//   lane = 16*half + n (column u = ut*16 + n), element e -> K = kt*32 + 8*half + (e&7) + 16*(e>>3)
// ---------------------------------------------------------------------------
__global__ void qrnn_pack_w(const float* __restrict__ Wz,
                            const float* __restrict__ Wf,
                            const float* __restrict__ Wo,
                            _Float16* __restrict__ wp) {
  int idx = blockIdx.x * blockDim.x + threadIdx.x;
  if (idx >= WPACK_HALFS) return;
  int e    = idx & 15;
  int t    = idx >> 4;
  int lane = t & 31;  t >>= 5;
  int g    = t % 3;   t /= 3;
  int kt   = t & 31;  t >>= 5;
  int ut   = t;
  int u    = ut * 16 + (lane & 15);
  int half = lane >> 4;
  int k    = kt * 32 + 8 * half + (e & 7) + 16 * (e >> 3);  // row of the 1024x512 weight
  const float* W = (g == 0) ? Wz : (g == 1) ? Wf : Wo;
  // (W,D,U) flat: [(w*ND + d)*NU + u] == [k*NU + u] with k = w*ND + d
  wp[idx] = (_Float16)W[(size_t)k * NU + u];
}

// ---------------------------------------------------------------------------
// Prologue 2: convert x to f16 with one zero row prepended per batch:
//   x16[b][0][:]   = 0            (the causal t = -1 pad)
//   x16[b][r][:]   = x[b][r-1][:] for r = 1..NT
// Each thread produces 8 contiguous halfs (one 16-byte chunk).
// ---------------------------------------------------------------------------
__global__ void qrnn_pack_x(const float* __restrict__ x,
                            _Float16* __restrict__ x16) {
  int i = blockIdx.x * blockDim.x + threadIdx.x;
  const int total = NB * (NT + 1) * (ND / 8);
  if (i >= total) return;
  int c8  = i & 63;            // ND/8 = 64 chunks per row
  int row = i >> 6;
  int b   = row / (NT + 1);
  int r   = row - b * (NT + 1);
  union { _Float16 h8[8]; uint4 u4; } pk;
  if (r == 0) {
    pk.u4 = make_uint4(0u, 0u, 0u, 0u);
  } else {
    const float* src = x + ((size_t)b * NT + (r - 1)) * ND + c8 * 8;
    float4 v0 = *(const float4*)(src);
    float4 v1 = *(const float4*)(src + 4);
    pk.h8[0] = (_Float16)v0.x; pk.h8[1] = (_Float16)v0.y;
    pk.h8[2] = (_Float16)v0.z; pk.h8[3] = (_Float16)v0.w;
    pk.h8[4] = (_Float16)v1.x; pk.h8[5] = (_Float16)v1.y;
    pk.h8[6] = (_Float16)v1.z; pk.h8[7] = (_Float16)v1.w;
  }
  *(uint4*)(x16 + (size_t)row * ND + c8 * 8) = pk.u4;
}

// Branch-free activations: ~1 ulp v_exp_f32 / v_rcp_f32, no exec divergence.
__device__ __forceinline__ float sigf(float v) {
  return __builtin_amdgcn_rcpf(1.0f + __expf(-v));
}
__device__ __forceinline__ float tanh_fast(float v) {
  // 2*sigmoid(2v) - 1 ; saturates correctly at +-1 for large |v|
  return 2.0f * __builtin_amdgcn_rcpf(1.0f + __expf(-2.0f * v)) - 1.0f;
}

__global__ void __launch_bounds__(NWAVES * 32, 1)
qrnn_main(const _Float16* __restrict__ x16,
          const _Float16* __restrict__ wp,
          const float* __restrict__ bz,
          const float* __restrict__ bf,
          const float* __restrict__ bo,
          float* __restrict__ out) {
  __shared__ _Float16 xs[XROWS * XSTR];                 // ~67.6 KB
  __shared__ float sa[NWAVES][CHUNK * SSTR];            // forget gate a
  __shared__ float sb[NWAVES][CHUNK * SSTR];            // (1-f)*z
  __shared__ float so[NWAVES][CHUNK * SSTR];            // o, then o*h

  const int b    = blockIdx.y;
  const int tid  = threadIdx.x;
  const int wv   = tid >> 5;
  const int lane = tid & 31;
  const int half = lane >> 4;
  const int ln   = lane & 15;
  const int ut   = blockIdx.x * NWAVES + wv;
  const int u0   = ut * 16;

  const float bzv = bz[u0 + ln];
  const float bfv = bf[u0 + ln];
  const float bov = bo[u0 + ln];

  const _Float16* wbase = wp + (size_t)ut * (NKT * 3 * 32 * 16);
  // x16 rows for this batch; global row (t0 + rr) == time t0 - 1 + rr
  const _Float16* xg = x16 + (size_t)b * (NT + 1) * ND;
  // LDS byte address of xs: low 32 bits of the flat shared-aperture pointer
  const unsigned ldsbase = (unsigned)(size_t)(void*)&xs[0];

  float h = 0.0f;  // recurrence carry, owned by lanes 0..15 (column ln)

  for (int t0 = 0; t0 < NT; t0 += CHUNK) {
    // ---- async-stage x16 rows [t0 .. t0+CHUNK] into LDS (16 B per lane) ----
    for (int i = tid; i < XROWS * (ND / 8); i += NWAVES * 32) {
      int rr = i >> 6;                 // row in tile (ND/8 = 64 chunks/row)
      int c  = i & 63;
      unsigned lds  = ldsbase + (unsigned)(rr * (XSTR * 2) + c * 16);
      unsigned voff = (unsigned)(((t0 + rr) * ND + c * 8) * 2);   // bytes
      asm volatile("global_load_async_to_lds_b128 %0, %1, %2"
                   :: "v"(lds), "v"(voff), "s"(xg)
                   : "memory");
    }
    asm volatile("s_wait_asynccnt 0x0" ::: "memory");
    __syncthreads();

    // ---- 3 gates x MT tiles of 16x16, K = 1024 via 32 k-tiles ----
    v8f zf = {};
    v8f acc[3][MT];
#pragma unroll
    for (int g = 0; g < 3; ++g)
#pragma unroll
      for (int m = 0; m < MT; ++m) acc[g][m] = zf;

#pragma unroll 1
    for (int kt = 0; kt < NKT; ++kt) {
      // A fragment: row M = ln, K-local = half*8 + {0..7, 16..23}
      // k-tiles 0..15 -> conv tap w=0 (row t-1), 16..31 -> w=1 (row t)
      const int rofs  = (kt >= 16) ? 1 : 0;
      const int dbase = (kt & 15) * 32 + half * 8;
      v16h afr[MT];
#pragma unroll
      for (int m = 0; m < MT; ++m) {
        const _Float16* ap = &xs[(m * 16 + ln + rofs) * XSTR + dbase];
        v8h lo = *(const v8h*)ap;
        v8h hi = *(const v8h*)(ap + 16);
        afr[m] = __builtin_shufflevector(lo, hi, 0, 1, 2, 3, 4, 5, 6, 7,
                                         8, 9, 10, 11, 12, 13, 14, 15);
      }
      const _Float16* wk = wbase + ((size_t)(kt * 3) * 32 + lane) * 16;
      if (kt + 1 < NKT) __builtin_prefetch(wk + 3 * 32 * 16, 0, 3);
#pragma unroll
      for (int g = 0; g < 3; ++g) {
        v16h bfr = *(const v16h*)(wk + (size_t)g * (32 * 16));
#pragma unroll
        for (int m = 0; m < MT; ++m) {
          acc[g][m] = __builtin_amdgcn_wmma_f32_16x16x32_f16(
              false, afr[m], false, bfr, (short)0, acc[g][m], false, false);
        }
      }
    }

    // ---- activations -> scan buffers ----
    // C/D layout: VGPR r holds M = r + 8*half, column N = ln
#pragma unroll
    for (int m = 0; m < MT; ++m) {
#pragma unroll
      for (int r = 0; r < 8; ++r) {
        int tl = m * 16 + half * 8 + r;
        float zv = tanh_fast(acc[0][m][r] + bzv);
        float fv = sigf(acc[1][m][r] + bfv);
        float ov = sigf(acc[2][m][r] + bov);
        sa[wv][tl * SSTR + ln] = fv;
        sb[wv][tl * SSTR + ln] = (1.0f - fv) * zv;
        so[wv][tl * SSTR + ln] = ov;
      }
    }
    __syncthreads();

    // ---- sequential recurrence: h_t = f*h + (1-f)*z; out = o*h ----
    if (lane < 16) {
      float hh = h;
#pragma unroll 4
      for (int i = 0; i < CHUNK; ++i) {
        float av = sa[wv][i * SSTR + ln];
        float bv = sb[wv][i * SSTR + ln];
        float ov = so[wv][i * SSTR + ln];
        hh = fmaf(av, hh, bv);
        so[wv][i * SSTR + ln] = ov * hh;  // stage output
      }
      h = hh;
    }
    __syncthreads();

    // ---- coalesced store: half-wave per row ----
    float* op = out + ((size_t)b * NT + t0) * NU + u0 + ln;
#pragma unroll 4
    for (int i = 0; i < CHUNK / 2; ++i) {
      int tl = i * 2 + half;
      op[(size_t)tl * NU] = so[wv][tl * SSTR + ln];
    }
    __syncthreads();
  }
}

extern "C" void kernel_launch(void* const* d_in, const int* in_sizes, int n_in,
                              void* d_out, int out_size, void* d_ws, size_t ws_size,
                              hipStream_t stream) {
  const float* x  = (const float*)d_in[0];
  const float* Wz = (const float*)d_in[1];
  const float* bz = (const float*)d_in[2];
  const float* Wf = (const float*)d_in[3];
  const float* bf = (const float*)d_in[4];
  const float* Wo = (const float*)d_in[5];
  const float* bo = (const float*)d_in[6];
  float* out = (float*)d_out;

  _Float16* wp  = (_Float16*)d_ws;                 // 3 MiB packed f16 weights
  _Float16* x16 = wp + WPACK_HALFS;                // 16*(4096+1)*512 halfs (~64 MiB)

  qrnn_pack_w<<<(WPACK_HALFS + 255) / 256, 256, 0, stream>>>(Wz, Wf, Wo, wp);

  const int xtotal = NB * (NT + 1) * (ND / 8);
  qrnn_pack_x<<<(xtotal + 255) / 256, 256, 0, stream>>>(x, x16);

  dim3 grid(NUT / NWAVES, NB);   // (16, 16) blocks
  qrnn_main<<<grid, NWAVES * 32, 0, stream>>>(x16, wp, bz, bf, bo, out);
}
